// attn_block_66357244723391
// MI455X (gfx1250) — matrix operand order repaired and checked by reading, not compiled, not run on hardware
//
#include <hip/hip_runtime.h>
#include <hip/hip_bf16.h>
#include <math.h>

typedef __attribute__((ext_vector_type(16))) _Float16 v16h;
typedef __attribute__((ext_vector_type(8)))  _Float16 v8h;
typedef __attribute__((ext_vector_type(4)))  _Float16 v4h;
typedef __attribute__((ext_vector_type(8)))  float    v8f;
typedef __attribute__((ext_vector_type(4)))  float    v4f;
typedef __attribute__((ext_vector_type(4)))  unsigned v4u;
typedef __attribute__((ext_vector_type(8)))  unsigned v8u;

#define B_SZ 8
#define T_SZ 2048
#define D_SZ 512

__device__ __forceinline__ v8f zero8() {
    v8f z;
#pragma unroll
    for (int i = 0; i < 8; i++) z[i] = 0.0f;
    return z;
}

__device__ __forceinline__ v16h combine16(v8h a, v8h b) {
    return __builtin_shufflevector(a, b, 0,1,2,3,4,5,6,7,8,9,10,11,12,13,14,15);
}

// 16 contiguous halves starting at p (16B-aligned)
__device__ __forceinline__ v16h ld16(const _Float16* p) {
    v8h a = *(const v8h*)p;
    v8h b = *(const v8h*)(p + 8);
    return combine16(a, b);
}

// two 8-half runs (A-operand layout helper)
__device__ __forceinline__ v16h ld8x2(const _Float16* p0, const _Float16* p1) {
    v8h a = *(const v8h*)p0;
    v8h b = *(const v8h*)p1;
    return combine16(a, b);
}

__device__ __forceinline__ float sigmoidf(float x) { return 1.0f / (1.0f + expf(-x)); }

// CDNA5 async global->LDS copy, 16B per lane, tracked by ASYNCcnt.
__device__ __forceinline__ void async_ld16(void* lds, const void* gaddr) {
    unsigned l = (unsigned)(size_t)lds;
    asm volatile("global_load_async_to_lds_b128 %0, %1, off"
                 :: "v"(l), "v"(gaddr)
                 : "memory");
}
__device__ __forceinline__ void wait_async0() {
    asm volatile("s_wait_asynccnt 0" ::: "memory");
}

// CDNA5 Tensor Data Mover: one wave-level DMA of a 2D tile into LDS with
// hardware row padding. Tracked by TENSORcnt.
__device__ __forceinline__ void tdm_load_2d(unsigned lds_addr, const void* gaddr,
                                            unsigned g1w0, unsigned tdim0,
                                            unsigned tdim1, unsigned tile0,
                                            unsigned tile1, unsigned stride0) {
    unsigned long long ga = (unsigned long long)(size_t)gaddr;
    v4u g0;
    g0[0] = 1u;                                   // count=1, user descriptor
    g0[1] = lds_addr;                             // LDS byte address
    g0[2] = (unsigned)ga;                         // global addr [31:0]
    g0[3] = (unsigned)(ga >> 32) | 0x80000000u;   // addr [56:32] | type=2
    v8u g1;
    g1[0] = g1w0;                // wg_mask=0 | data_size | pad cfg
    g1[1] = (tdim0 & 0xFFFFu) << 16;              // tensor_dim0 [79:48] lo
    g1[2] = (tdim0 >> 16) | ((tdim1 & 0xFFFFu) << 16);   // + tensor_dim1 lo
    g1[3] = (tdim1 >> 16) | (tile0 << 16);        // + tile_dim0
    g1[4] = tile1;                                // tile_dim1 (tile_dim2=0)
    g1[5] = stride0;                              // tensor_dim0_stride [31:0]
    g1[6] = 0u;                                   // stride hi, dim1_stride lo
    g1[7] = 0u;
    asm volatile("tensor_load_to_lds %0, %1" :: "s"(g0), "s"(g1) : "memory");
}

// ---------------------------------------------------------------------------
// Kernel 1: per-batch gate vectors.
// ---------------------------------------------------------------------------
__global__ void __launch_bounds__(256) gates_kernel(
    const float* __restrict__ aux, const float* __restrict__ Wvs,
    const float* __restrict__ bvs, const float* __restrict__ Wvt,
    const float* __restrict__ bvt, float* __restrict__ sq,
    float* __restrict__ sk, float* __restrict__ gv)
{
    int idx = blockIdx.x * blockDim.x + threadIdx.x;   // 0..B*D-1
    int b = idx >> 9;
    int j = idx & (D_SZ - 1);
    const float* a = aux + (size_t)b * (3 * D_SZ);
    sq[idx] = sigmoidf(a[j]);
    sk[idx] = sigmoidf(a[D_SZ + j]);
    const float* ws = Wvs + (size_t)j * D_SZ;
    const float* wt = Wvt + (size_t)j * D_SZ;
    float accs = 0.0f, acct = 0.0f;
    for (int d = 0; d < D_SZ; d++) {
        float sv = sigmoidf(a[2 * D_SZ + d]);
        accs += sv * ws[d];
        acct += sv * wt[d];
    }
    gv[idx] = sigmoidf(accs + bvs[j]) * tanhf(acct + bvt[j]);
}

// ---------------------------------------------------------------------------
// Kernel 2a: Q projection GEMM via WMMA.
//   qp[b,t,j] = ((query[b,t,:]·Wq[j,:]) + bq[j]) * sq[b,j]   -> f16
// ---------------------------------------------------------------------------
__global__ void __launch_bounds__(32) qproj_kernel(
    const float* __restrict__ query, const float* __restrict__ Wq,
    const float* __restrict__ bq, const float* __restrict__ sq,
    _Float16* __restrict__ qp)
{
    int b  = blockIdx.z;
    int t0 = blockIdx.x * 16;
    int jb = blockIdx.y * 128;
    int lane = threadIdx.x & 31;
    int ln = lane & 15, lg = lane >> 4;

    const float* qr = query + ((size_t)b * T_SZ + t0 + ln) * D_SZ;

    v8f acc[8];
#pragma unroll
    for (int jt = 0; jt < 8; jt++) acc[jt] = zero8();

#pragma unroll
    for (int c = 0; c < 16; c++) {
        v8f a0 = *(const v8f*)(qr + 32 * c + lg * 8);
        v8f a1 = *(const v8f*)(qr + 32 * c + 16 + lg * 8);
        v16h af = combine16(__builtin_convertvector(a0, v8h),
                            __builtin_convertvector(a1, v8h));
#pragma unroll
        for (int jt = 0; jt < 8; jt++) {
            const float* wr = Wq + (size_t)(jb + jt * 16 + ln) * D_SZ + lg * 8 + 32 * c;
            v8f b0 = *(const v8f*)wr;
            v8f b1 = *(const v8f*)(wr + 16);
            v16h bf = combine16(__builtin_convertvector(b0, v8h),
                                __builtin_convertvector(b1, v8h));
            acc[jt] = __builtin_amdgcn_wmma_f32_16x16x32_f16(
                false, af, false, bf, (short)0, acc[jt], false, false);
        }
    }
#pragma unroll
    for (int jt = 0; jt < 8; jt++) {
        int j = jb + jt * 16 + ln;
        float bias = bq[j];
        float g = sq[(size_t)b * D_SZ + j];
#pragma unroll
        for (int i = 0; i < 8; i++) {
            int row = t0 + i + 8 * lg;
            qp[((size_t)b * T_SZ + row) * D_SZ + j] = (_Float16)((acc[jt][i] + bias) * g);
        }
    }
}

// ---------------------------------------------------------------------------
// Kernel 2b: elementwise gating + layout packing.
// ---------------------------------------------------------------------------
__global__ void __launch_bounds__(256) gate_kv_kernel(
    const float* __restrict__ value, const float* __restrict__ sk,
    const float* __restrict__ gv, _Float16* __restrict__ kk,
    _Float16* __restrict__ vvT)
{
    __shared__ __align__(16) _Float16 lt[64][72];
    int b  = blockIdx.z;
    int s0 = blockIdx.x * 64;
    int d0 = blockIdx.y * 64;
    int tid = threadIdx.x;
    int cx = tid & 15, ry = tid >> 4;
    int dbase = d0 + cx * 4;

    v4f skv = *(const v4f*)(sk + (size_t)b * D_SZ + dbase);
    v4f gvv = *(const v4f*)(gv + (size_t)b * D_SZ + dbase);

#pragma unroll
    for (int it = 0; it < 4; it++) {
        int s = s0 + ry + it * 16;
        v4f val = *(const v4f*)(value + ((size_t)b * T_SZ + s) * D_SZ + dbase);
        v4f kg = val * skv;
        v4f vg = val * gvv;
        *(v4h*)(kk + ((size_t)b * T_SZ + s) * D_SZ + dbase) = __builtin_convertvector(kg, v4h);
        *(v4h*)(&lt[ry + it * 16][cx * 4]) = __builtin_convertvector(vg, v4h);
    }
    __syncthreads();
#pragma unroll
    for (int it = 0; it < 4; it++) {
        int dl = ry + it * 16;
        v4h o;
#pragma unroll
        for (int k2 = 0; k2 < 4; k2++) o[k2] = lt[cx * 4 + k2][dl];
        *(v4h*)(vvT + ((size_t)b * D_SZ + d0 + dl) * T_SZ + s0 + cx * 4) = o;
    }
}

// ---------------------------------------------------------------------------
// Kernel 3: flash attention. K' tile staged by the Tensor Data Mover
// (TENSORcnt, hardware LDS padding), V'^T tile by per-lane async loads
// (ASYNCcnt). WMMA f16 with f32 accumulation.
// ---------------------------------------------------------------------------
#define KPITCH 520   // 512 + 8 halves (TDM pad: 4 dwords every 256 dwords)
#define VPITCH 72    // 64 + 8 halves
#define PPITCH 72

__global__ void __launch_bounds__(128) fa_kernel(
    const _Float16* __restrict__ qp, const _Float16* __restrict__ kk,
    const _Float16* __restrict__ vvT, float* __restrict__ out)
{
    extern __shared__ _Float16 sm[];
    _Float16* kb = sm;                                  // 64 x KPITCH
    _Float16* vb = sm + 64 * KPITCH;                    // 512 x VPITCH
    _Float16* pball = sm + 64 * KPITCH + 512 * VPITCH;  // 4 x 16 x PPITCH

    int b  = blockIdx.y;
    int m0 = blockIdx.x * 64;
    int tid  = threadIdx.x;
    int wave = tid >> 5;
    int lane = tid & 31;
    int ln = lane & 15, lg = lane >> 4;
    int mrow0 = m0 + wave * 16;
    _Float16* pb = pball + wave * (16 * PPITCH);

    const float NEG_INF = -__builtin_inff();
    const float cscale = 0.044194173824159216f * 1.4426950408889634f; // /sqrt(D)*log2e

    // Persistent Q fragments (A-operand layout).
    v16h qf[16];
    {
        const _Float16* qrow = qp + ((size_t)b * T_SZ + mrow0 + ln) * D_SZ + lg * 8;
#pragma unroll
        for (int c = 0; c < 16; c++)
            qf[c] = ld8x2(qrow + 32 * c, qrow + 32 * c + 16);
    }

    v8f o[32];
#pragma unroll
    for (int i = 0; i < 32; i++) o[i] = zero8();
    float mrow[8], lrow[8];
#pragma unroll
    for (int i = 0; i < 8; i++) { mrow[i] = NEG_INF; lrow[i] = 0.0f; }

    unsigned kb_lds = (unsigned)(size_t)kb;

    int nb = m0 / 64 + 1;                 // causal: s-blocks 0..m0/64
    for (int sb = 0; sb < nb; sb++) {
        int s0 = sb * 64;
        bool diag = (sb == nb - 1);

        __syncthreads();  // previous iteration's LDS readers are done

        // ---- K' tile (64 rows x 1024B) via TDM, one issue from wave 0.
        // data_size=4B(2); pad_enable, pad_interval=7 (256 dw), pad_amount=3 (4 dw)
        if (wave == 0) {
            const _Float16* kg = kk + ((size_t)b * T_SZ + s0) * D_SZ;
            tdm_load_2d(kb_lds, kg,
                        (2u << 16) | (1u << 20) | (7u << 22) | (3u << 25),
                        /*tensor_dim0=*/256u, /*tensor_dim1=*/(unsigned)(T_SZ - s0),
                        /*tile0=*/256u, /*tile1=*/64u, /*stride0=*/256u);
        }
        // ---- V'^T tile (512 rows x 128B) via per-lane async loads.
        {
            const _Float16* vg = vvT + (size_t)b * D_SZ * T_SZ + s0;
#pragma unroll 4
            for (int i = 0; i < 32; i++) {
                int idx = tid + 128 * i;
                int d = idx >> 3, c = idx & 7;
                async_ld16(vb + d * VPITCH + c * 8, vg + (size_t)d * T_SZ + c * 8);
            }
        }
        wait_async0();                                   // this wave's async writes
        if (wave == 0) __builtin_amdgcn_s_wait_tensorcnt(0);  // TDM complete
        __syncthreads();                                 // all LDS writes visible

        // ---- scores S = Q·K'^T over full D=512
        float wl[4][8];
#pragma unroll
        for (int nt = 0; nt < 4; nt++) {
            v8f acc = zero8();
            const _Float16* kr = kb + (size_t)(nt * 16 + ln) * KPITCH + lg * 8;
#pragma unroll
            for (int c = 0; c < 16; c++)
                acc = __builtin_amdgcn_wmma_f32_16x16x32_f16(
                    false, qf[c], false, ld8x2(kr + 32 * c, kr + 32 * c + 16), (short)0, acc, false, false);
            int col = s0 + nt * 16 + ln;
#pragma unroll
            for (int i = 0; i < 8; i++) {
                int row = mrow0 + i + 8 * lg;
                float v = acc[i];
                bool kill = (v == 0.0f) || (diag && (col > row));  // faithful tril*w==0 -> -inf
                wl[nt][i] = kill ? NEG_INF : v * cscale;
            }
        }

        // ---- online softmax row statistics (butterfly over 16 lanes)
        float scr[8];
#pragma unroll
        for (int i = 0; i < 8; i++) {
            float tm = fmaxf(fmaxf(wl[0][i], wl[1][i]), fmaxf(wl[2][i], wl[3][i]));
#pragma unroll
            for (int off = 1; off < 16; off <<= 1)
                tm = fmaxf(tm, __shfl_xor(tm, off, 32));
            float mn = fmaxf(mrow[i], tm);
            scr[i] = exp2f(mrow[i] - mn);
            mrow[i] = mn;
            float rs = 0.0f;
#pragma unroll
            for (int nt = 0; nt < 4; nt++) {
                float p = exp2f(wl[nt][i] - mn);
                wl[nt][i] = p;
                rs += p;
            }
#pragma unroll
            for (int off = 1; off < 16; off <<= 1)
                rs += __shfl_xor(rs, off, 32);
            lrow[i] = lrow[i] * scr[i] + rs;
        }

        // ---- P -> f16 via per-wave LDS bounce (C layout -> A-operand layout)
#pragma unroll
        for (int nt = 0; nt < 4; nt++)
#pragma unroll
            for (int i = 0; i < 8; i++)
                pb[(size_t)(i + 8 * lg) * PPITCH + nt * 16 + ln] = (_Float16)wl[nt][i];

        const _Float16* pr = pb + (size_t)ln * PPITCH + lg * 16;
        v16h pf0 = ld16(pr);                  // K = 0..31
        v16h pf1 = ld16(pr + 32);             // K = 32..63

        // ---- O = O*scale + P·V'
#pragma unroll
        for (int dt = 0; dt < 32; dt++) {
            const _Float16* vr = vb + (size_t)(dt * 16 + ln) * VPITCH + lg * 16;
            v16h vf0 = ld16(vr);
            v16h vf1 = ld16(vr + 32);
            v8f acc = o[dt];
#pragma unroll
            for (int i = 0; i < 8; i++) acc[i] *= scr[i];
            acc = __builtin_amdgcn_wmma_f32_16x16x32_f16(
                false, pf0, false, vf0, (short)0, acc, false, false);
            acc = __builtin_amdgcn_wmma_f32_16x16x32_f16(
                false, pf1, false, vf1, (short)0, acc, false, false);
            o[dt] = acc;
        }
    }

    // ---- epilogue: out = O / l
    float inv[8];
#pragma unroll
    for (int i = 0; i < 8; i++) inv[i] = 1.0f / lrow[i];
    float* ob = out + (size_t)b * T_SZ * D_SZ;
#pragma unroll
    for (int dt = 0; dt < 32; dt++) {
        int d = dt * 16 + ln;
#pragma unroll
        for (int i = 0; i < 8; i++) {
            int row = mrow0 + i + 8 * lg;
            ob[(size_t)row * D_SZ + d] = o[dt][i] * inv[i];
        }
    }
}

// ---------------------------------------------------------------------------
extern "C" void kernel_launch(void* const* d_in, const int* in_sizes, int n_in,
                              void* d_out, int out_size, void* d_ws, size_t ws_size,
                              hipStream_t stream) {
    const float* query = (const float*)d_in[0];
    const float* value = (const float*)d_in[1];
    const float* aux   = (const float*)d_in[2];
    const float* Wq    = (const float*)d_in[3];
    const float* bq    = (const float*)d_in[4];
    const float* Wvs   = (const float*)d_in[5];
    const float* bvs   = (const float*)d_in[6];
    const float* Wvt   = (const float*)d_in[7];
    const float* bvt   = (const float*)d_in[8];
    float* out = (float*)d_out;

    char* ws = (char*)d_ws;
    float* sq = (float*)ws;                 // B*D f32
    float* sk = sq + B_SZ * D_SZ;           // B*D f32
    float* gv = sk + B_SZ * D_SZ;           // B*D f32
    _Float16* qp  = (_Float16*)(ws + 65536);
    _Float16* kk  = qp + (size_t)B_SZ * T_SZ * D_SZ;
    _Float16* vvT = kk + (size_t)B_SZ * T_SZ * D_SZ;

    gates_kernel<<<dim3((B_SZ * D_SZ) / 256), 256, 0, stream>>>(
        aux, Wvs, bvs, Wvt, bvt, sq, sk, gv);

    qproj_kernel<<<dim3(T_SZ / 16, D_SZ / 128, B_SZ), 32, 0, stream>>>(
        query, Wq, bq, sq, qp);

    gate_kv_kernel<<<dim3(T_SZ / 64, D_SZ / 64, B_SZ), 256, 0, stream>>>(
        value, sk, gv, kk, vvT);

    const int smem = (64 * KPITCH + 512 * VPITCH + 4 * 16 * PPITCH) * 2;  // 149504 B
    (void)hipFuncSetAttribute((const void*)fa_kernel,
                              hipFuncAttributeMaxDynamicSharedMemorySize, smem);
    fa_kernel<<<dim3(T_SZ / 64, B_SZ), 128, smem, stream>>>(qp, kk, vvT, out);
}
